// T5EncoderBlock_53738630808204
// MI455X (gfx1250) — matrix operand order repaired and checked by reading, not compiled, not run on hardware
//
#include <hip/hip_runtime.h>
#include <hip/hip_bf16.h>
#include <math.h>
#include <stdint.h>

// ---------------------------------------------------------------------------
// Types for CDNA5 WMMA (wave32): V_WMMA_F32_16X16X32_BF16
// ---------------------------------------------------------------------------
typedef __bf16 bf16_t;
typedef __attribute__((ext_vector_type(16))) __bf16 v16bf;
typedef __attribute__((ext_vector_type(8)))  __bf16 bf16x8;
typedef __attribute__((ext_vector_type(8)))  float  v8f;

union AB16 { v16bf v; bf16x8 h[2]; };

__device__ __forceinline__ v8f wmma_bf16(const AB16& a, const AB16& b, v8f c) {
  // 8 args: (neg_a, A, neg_b, B, c_mod, C, reuse_a, reuse_b)
  return __builtin_amdgcn_wmma_f32_16x16x32_bf16(false, a.v, false, b.v,
                                                 (short)0, c, false, false);
}

// ---------------------------------------------------------------------------
// CDNA5 async global->LDS copy (GLOBAL_LOAD_ASYNC_TO_LDS_B128, ASYNCcnt).
// LDS address = low 32 bits of the generic shared pointer (ISA 10.2 aperture
// rule); global address in a 64-bit VGPR pair, GV mode (saddr = off).
// ---------------------------------------------------------------------------
__device__ __forceinline__ void async_copy_b128(void* lds, const void* gptr) {
  unsigned       l = (unsigned)(uintptr_t)lds;
  unsigned long long g = (unsigned long long)(uintptr_t)gptr;
  asm volatile("global_load_async_to_lds_b128 %0, %1, off"
               :: "v"(l), "v"(g) : "memory");
}
__device__ __forceinline__ void wait_async0() {
  asm volatile("s_wait_asynccnt 0x0" ::: "memory");
}

// ---------------------------------------------------------------------------
// f32 -> bf16 cast (activations)
// ---------------------------------------------------------------------------
__global__ void cast_f32_bf16(const float* __restrict__ in,
                              bf16_t* __restrict__ out, int n) {
  int i = blockIdx.x * blockDim.x + threadIdx.x;
  const int stride = gridDim.x * blockDim.x;
  for (; i < n; i += stride) out[i] = (bf16_t)in[i];
}

// ---------------------------------------------------------------------------
// Fused f32 -> bf16 cast + transpose for weights:
//   in:  [K][N] f32 (row-major)   out: [N][K] bf16 (row-major)
// ---------------------------------------------------------------------------
__global__ __launch_bounds__(256)
void cast_transpose_f32_bf16(const float* __restrict__ in, bf16_t* __restrict__ out,
                             int K, int N) {
  __shared__ float tile[32][33];
  const int k0 = blockIdx.y * 32, n0 = blockIdx.x * 32;
  const int tx = threadIdx.x & 31, ty = threadIdx.x >> 5;  // 32 x 8
#pragma unroll
  for (int i = ty; i < 32; i += 8)
    tile[i][tx] = in[(size_t)(k0 + i) * N + n0 + tx];
  __syncthreads();
#pragma unroll
  for (int i = ty; i < 32; i += 8)
    out[(size_t)(n0 + i) * K + k0 + tx] = (bf16_t)tile[tx][i];
}

// ---------------------------------------------------------------------------
// bf16 WMMA GEMM: C[M,N] = A[M,K] @ B[K,N] + bias, B given PRE-TRANSPOSED
// as Bt[N][K]. Epilogues:
//   EPI=0: store bf16(C)                (QKV projections)
//   EPI=1: store f32(C + resid)         (attn out-proj, FFN2 -> pre-norm)
//   EPI=2: store bf16(gelu_exact(C))    (FFN1)
// Block: 256 threads (8 waves). Tile 128(M) x 256(N), K-step 32,
// double-buffered LDS staged via async global->LDS DMA. Wave grid 2x4; each
// wave owns 64x64 = 4x4 WMMA frags -> 16 ds_load_b128 : 16 v_wmma per k-step.
// ---------------------------------------------------------------------------
template<int EPI>
__global__ __launch_bounds__(256)
void gemm_bf16_kernel(const bf16_t* __restrict__ A, const bf16_t* __restrict__ Bt,
                      const float* __restrict__ bias, const float* __restrict__ resid,
                      float* __restrict__ outF, bf16_t* __restrict__ outB,
                      int M, int N, int K)
{
  constexpr int BM = 128, BN = 256, BK = 32;
  __shared__ __align__(16) bf16_t As[2][BM * BK];   // [row][k]
  __shared__ __align__(16) bf16_t Bs[2][BN * BK];   // [n][k]

  const int tid  = threadIdx.x;
  const int wave = tid >> 5, lane = tid & 31;
  const int wm = wave >> 2, wn = wave & 3;          // 2 x 4 wave grid
  const int n16 = lane & 15, hl = lane >> 4;
  const int m0 = blockIdx.y * BM, n0 = blockIdx.x * BN;

  v8f acc[4][4];
#pragma unroll
  for (int mi = 0; mi < 4; ++mi)
#pragma unroll
    for (int ni = 0; ni < 4; ++ni)
#pragma unroll
      for (int e = 0; e < 8; ++e) acc[mi][ni][e] = 0.f;

  // Staging map: A: thread t -> (row t>>1, 16-elem seg (t&1)*16), 2x b128.
  //              B: thread t -> row n = t, 32 elems = 4x b128.
  const int ar = tid >> 1;
  const int ac = (tid & 1) * 16;
  const bf16_t* gA = A  + (size_t)(m0 + ar) * K + ac;
  const bf16_t* gB = Bt + (size_t)(n0 + tid) * K;

  auto stage = [&](int buf, int kk) {
    async_copy_b128(As[buf] + ar * BK + ac,     gA + kk);
    async_copy_b128(As[buf] + ar * BK + ac + 8, gA + kk + 8);
    bf16_t* lb = Bs[buf] + tid * BK;
#pragma unroll
    for (int e = 0; e < 4; ++e)
      async_copy_b128(lb + 8 * e, gB + kk + 8 * e);
  };

  stage(0, 0);
  for (int kk = 0; kk < K; kk += BK) {
    const int buf = (kk >> 5) & 1;
    wait_async0();       // this wave's DMA into `buf` is complete
    __syncthreads();     // ... and every other wave's too
    if (kk + BK < K) stage(buf ^ 1, kk + BK);   // overlaps with WMMAs below

    AB16 af[4], bfr[4];
#pragma unroll
    for (int mi = 0; mi < 4; ++mi) {
      const bf16_t* p = As[buf] + (wm * 64 + mi * 16 + n16) * BK;
      af[mi].h[0] = *(const bf16x8*)(p + 8 * hl);        // K = 8h..8h+7
      af[mi].h[1] = *(const bf16x8*)(p + 16 + 8 * hl);   // K = 16+8h..
    }
#pragma unroll
    for (int ni = 0; ni < 4; ++ni) {
      const bf16_t* p = Bs[buf] + (wn * 64 + ni * 16 + n16) * BK;
      bfr[ni].h[0] = *(const bf16x8*)(p + 8 * hl);        // K = 8h..8h+7
      bfr[ni].h[1] = *(const bf16x8*)(p + 16 + 8 * hl);   // K = 16+8h..
    }
#pragma unroll
    for (int mi = 0; mi < 4; ++mi)
#pragma unroll
      for (int ni = 0; ni < 4; ++ni)
        acc[mi][ni] = wmma_bf16(af[mi], bfr[ni], acc[mi][ni]);
  }

  // Epilogue. C-layout: lane holds col n16, rows v + 8*hl of each 16x16 frag.
#pragma unroll
  for (int mi = 0; mi < 4; ++mi) {
#pragma unroll
    for (int v = 0; v < 8; ++v) {
      const int row = m0 + wm * 64 + mi * 16 + v + 8 * hl;
#pragma unroll
      for (int ni = 0; ni < 4; ++ni) {
        const int col = n0 + wn * 64 + ni * 16 + n16;
        float val = acc[mi][ni][v] + bias[col];
        const size_t idx = (size_t)row * N + col;
        if constexpr (EPI == 0) {
          outB[idx] = (bf16_t)val;
        } else if constexpr (EPI == 1) {
          outF[idx] = val + resid[idx];
        } else {
          outB[idx] = (bf16_t)(0.5f * val * (1.0f + erff(val * 0.70710678118f)));
        }
      }
    }
  }
}

// ---------------------------------------------------------------------------
// Flash attention with relative-position bias.
// Grid: (S/128, B*H). Block 256 = 8 waves; wave w owns q rows [128*bx+16w, +16).
// Loops over keys in chunks of 32; K staged via async DMA, V transposed in LDS.
// ---------------------------------------------------------------------------
__global__ __launch_bounds__(256)
void attn_kernel(const bf16_t* __restrict__ Q, const bf16_t* __restrict__ Kmat,
                 const bf16_t* __restrict__ Vmat, const float* __restrict__ rel_pos,
                 bf16_t* __restrict__ Out)
{
  constexpr int S = 2048, D = 1024, HD = 64, NREL = 65;
  const int qblk = blockIdx.x;
  const int b = blockIdx.y >> 4, h = blockIdx.y & 15;
  const int tid = threadIdx.x, wave = tid >> 5, lane = tid & 31;
  const int n16 = lane & 15, hl = lane >> 4;

  __shared__ __align__(16) bf16_t Ks[32 * HD];    // [key][hd] row-major
  __shared__ __align__(16) bf16_t Vt[HD * 32];    // [hd][key] transposed
  __shared__ __align__(16) bf16_t Ps[8][16 * 32]; // per-wave P scratch
  __shared__ float relb[NREL];

  if (tid < NREL) relb[tid] = rel_pos[h * NREL + tid];

  // Q fragments, loaded once directly from global (HD contiguous).
  const int qrow0 = qblk * 128 + wave * 16;
  const bf16_t* qptr = Q + (size_t)(b * S + qrow0 + n16) * D + h * HD;
  AB16 qa[2];
  qa[0].h[0] = *(const bf16x8*)(qptr + 8 * hl);
  qa[0].h[1] = *(const bf16x8*)(qptr + 16 + 8 * hl);
  qa[1].h[0] = *(const bf16x8*)(qptr + 32 + 8 * hl);
  qa[1].h[1] = *(const bf16x8*)(qptr + 48 + 8 * hl);

  float m[8], lsum[8];
  v8f o[4];
#pragma unroll
  for (int ni = 0; ni < 4; ++ni)
#pragma unroll
    for (int e = 0; e < 8; ++e) o[ni][e] = 0.f;
  int qrow[8];
#pragma unroll
  for (int v = 0; v < 8; ++v) { m[v] = -3.0e38f; lsum[v] = 0.f; qrow[v] = qrow0 + v + 8 * hl; }

  const int skey = tid >> 3;           // 0..31
  const int shs  = (tid & 7) * 8;      // 0..56

  for (int j0 = 0; j0 < S; j0 += 32) {
    __syncthreads();
    // K tile: async DMA straight into LDS. V tile: manual transpose.
    async_copy_b128(Ks + skey * HD + shs,
                    Kmat + (size_t)(b * S + j0 + skey) * D + h * HD + shs);
    bf16x8 vv = *(const bf16x8*)(Vmat + (size_t)(b * S + j0 + skey) * D + h * HD + shs);
#pragma unroll
    for (int e = 0; e < 8; ++e) Vt[(shs + e) * 32 + skey] = vv[e];
    wait_async0();
    __syncthreads();

    // Scores for two 16-key subtiles: S16 = Q(16x64) @ K^T(64x16)
    v8f sf[2];
#pragma unroll
    for (int t = 0; t < 2; ++t) {
      AB16 kb0, kb1;
      const bf16_t* kr = Ks + (t * 16 + n16) * HD;     // lane's key row
      kb0.h[0] = *(const bf16x8*)(kr + 8 * hl);
      kb0.h[1] = *(const bf16x8*)(kr + 16 + 8 * hl);
      kb1.h[0] = *(const bf16x8*)(kr + 32 + 8 * hl);
      kb1.h[1] = *(const bf16x8*)(kr + 48 + 8 * hl);
      v8f s;
#pragma unroll
      for (int e = 0; e < 8; ++e) s[e] = 0.f;
      s = wmma_bf16(qa[0], kb0, s);
      s = wmma_bf16(qa[1], kb1, s);
      sf[t] = s;
    }

    // Online softmax over the 32-key chunk (row groups of 16 lanes, wave32).
    const int key0 = j0 + n16, key1 = j0 + 16 + n16;
#pragma unroll
    for (int v = 0; v < 8; ++v) {
      int r0 = key0 - qrow[v]; r0 = r0 < -32 ? -32 : (r0 > 32 ? 32 : r0);
      int r1 = key1 - qrow[v]; r1 = r1 < -32 ? -32 : (r1 > 32 ? 32 : r1);
      float sc0 = sf[0][v] * 0.125f + relb[r0 + 32];
      float sc1 = sf[1][v] * 0.125f + relb[r1 + 32];
      float rmax = fmaxf(sc0, sc1);
      rmax = fmaxf(rmax, __shfl_xor(rmax, 1, 32));
      rmax = fmaxf(rmax, __shfl_xor(rmax, 2, 32));
      rmax = fmaxf(rmax, __shfl_xor(rmax, 4, 32));
      rmax = fmaxf(rmax, __shfl_xor(rmax, 8, 32));
      const float mn  = fmaxf(m[v], rmax);
      const float pe0 = __expf(sc0 - mn);
      const float pe1 = __expf(sc1 - mn);
      float rsum = pe0 + pe1;
      rsum += __shfl_xor(rsum, 1, 32);
      rsum += __shfl_xor(rsum, 2, 32);
      rsum += __shfl_xor(rsum, 4, 32);
      rsum += __shfl_xor(rsum, 8, 32);
      const float corr = __expf(m[v] - mn);
      lsum[v] = lsum[v] * corr + rsum;
      m[v] = mn;
#pragma unroll
      for (int ni = 0; ni < 4; ++ni) o[ni][v] *= corr;
      Ps[wave][(v + 8 * hl) * 32 + n16]      = (bf16_t)pe0;
      Ps[wave][(v + 8 * hl) * 32 + 16 + n16] = (bf16_t)pe1;
    }
    __syncthreads();  // make P visible for A-layout reload

    // O += P(16x32) @ V(32x64)
    AB16 pa;
    const bf16_t* pr = &Ps[wave][n16 * 32];
    pa.h[0] = *(const bf16x8*)(pr + 8 * hl);
    pa.h[1] = *(const bf16x8*)(pr + 16 + 8 * hl);
#pragma unroll
    for (int ni = 0; ni < 4; ++ni) {
      AB16 vb;
      const bf16_t* vr = Vt + (ni * 16 + n16) * 32;
      vb.h[0] = *(const bf16x8*)(vr + 8 * hl);
      vb.h[1] = *(const bf16x8*)(vr + 16 + 8 * hl);
      o[ni] = wmma_bf16(pa, vb, o[ni]);
    }
  }

  // Normalize and store (token-major [B*S, D], head slice contiguous).
#pragma unroll
  for (int v = 0; v < 8; ++v) {
    const float inv = 1.0f / lsum[v];
#pragma unroll
    for (int ni = 0; ni < 4; ++ni)
      Out[(size_t)(b * S + qrow[v]) * D + h * HD + ni * 16 + n16] =
          (bf16_t)(o[ni][v] * inv);
  }
}

// ---------------------------------------------------------------------------
// RMSNorm over rows of D floats: out = in / sqrt(mean(in^2)+eps) * g
// ---------------------------------------------------------------------------
__global__ __launch_bounds__(256)
void rmsnorm_kernel(const float* __restrict__ in, const float* __restrict__ g,
                    float* __restrict__ outF, bf16_t* __restrict__ outB, int D)
{
  const int row = blockIdx.x;
  const float* r = in + (size_t)row * D;
  float ss = 0.f;
  for (int c = threadIdx.x; c < D; c += 256) { float x = r[c]; ss += x * x; }
  __shared__ float red[256];
  red[threadIdx.x] = ss;
  __syncthreads();
  for (int off = 128; off > 0; off >>= 1) {
    if (threadIdx.x < off) red[threadIdx.x] += red[threadIdx.x + off];
    __syncthreads();
  }
  const float inv = rsqrtf(red[0] / (float)D + 1e-5f);
  for (int c = threadIdx.x; c < D; c += 256) {
    const float y = r[c] * inv * g[c];
    outF[(size_t)row * D + c] = y;
    if (outB) outB[(size_t)row * D + c] = (bf16_t)y;
  }
}

// ---------------------------------------------------------------------------
// Launch: cast/transpose -> QKV gemms -> attention -> out-proj(+res) ->
//         rmsnorm -> FFN1(gelu) -> FFN2(+res) -> rmsnorm -> d_out
// ---------------------------------------------------------------------------
extern "C" void kernel_launch(void* const* d_in, const int* in_sizes, int n_in,
                              void* d_out, int out_size, void* d_ws, size_t ws_size,
                              hipStream_t stream)
{
  constexpr int Bc = 4, Sc = 2048, Dc = 1024, Hc = 16, DFFc = 4096;
  constexpr int Mtok = Bc * Sc;  // 8192

  const float* x   = (const float*)d_in[0];
  const float* g1  = (const float*)d_in[1];
  const float* g2  = (const float*)d_in[2];
  const float* wq  = (const float*)d_in[3];
  const float* bq  = (const float*)d_in[4];
  const float* wk  = (const float*)d_in[5];
  const float* bkp = (const float*)d_in[6];
  const float* wv  = (const float*)d_in[7];
  const float* bv  = (const float*)d_in[8];
  const float* wo  = (const float*)d_in[9];
  const float* bo  = (const float*)d_in[10];
  const float* rel = (const float*)d_in[11];
  const float* w1  = (const float*)d_in[12];
  const float* b1  = (const float*)d_in[13];
  const float* w2  = (const float*)d_in[14];
  const float* b2  = (const float*)d_in[15];

  char* ws = (char*)d_ws;
  const size_t MBs = 1024ull * 1024ull;
  bf16_t* wqt  = (bf16_t*)(ws + 0   * MBs);  // [N][K] transposed bf16 weights
  bf16_t* wkt  = (bf16_t*)(ws + 2   * MBs);
  bf16_t* wvt  = (bf16_t*)(ws + 4   * MBs);
  bf16_t* wot  = (bf16_t*)(ws + 6   * MBs);
  bf16_t* w1t  = (bf16_t*)(ws + 8   * MBs);
  bf16_t* w2t  = (bf16_t*)(ws + 16  * MBs);
  bf16_t* xb   = (bf16_t*)(ws + 24  * MBs);
  bf16_t* qb   = (bf16_t*)(ws + 40  * MBs);
  bf16_t* kb   = (bf16_t*)(ws + 56  * MBs);
  bf16_t* vb   = (bf16_t*)(ws + 72  * MBs);
  bf16_t* aob  = (bf16_t*)(ws + 88  * MBs);
  float*  y1   = (float*)(ws + 104 * MBs);   // pre-norm1; reused for pre-norm2
  float*  y1n  = (float*)(ws + 136 * MBs);   // normalized (f32, FFN residual)
  bf16_t* y1nb = (bf16_t*)(ws + 168 * MBs);  // normalized (bf16, FFN input)
  bf16_t* hb   = (bf16_t*)(ws + 184 * MBs);  // GELU(FFN1) bf16, 64 MB

  {
    int nel = Mtok * Dc;
    int blocks = (nel + 256 * 8 - 1) / (256 * 8);
    if (blocks > 4096) blocks = 4096;
    cast_f32_bf16<<<blocks, 256, 0, stream>>>(x, xb, nel);
  }
  auto castT = [&](const float* s, bf16_t* dst, int K, int N) {
    cast_transpose_f32_bf16<<<dim3(N / 32, K / 32), 256, 0, stream>>>(s, dst, K, N);
  };
  castT(wq, wqt, Dc, Dc);
  castT(wk, wkt, Dc, Dc);
  castT(wv, wvt, Dc, Dc);
  castT(wo, wot, Dc, Dc);
  castT(w1, w1t, Dc, DFFc);
  castT(w2, w2t, DFFc, Dc);

  const dim3 blk(256);
  const dim3 gDD(Dc / 256, Mtok / 128);
  gemm_bf16_kernel<0><<<gDD, blk, 0, stream>>>(xb, wqt, bq,  nullptr, nullptr, qb, Mtok, Dc, Dc);
  gemm_bf16_kernel<0><<<gDD, blk, 0, stream>>>(xb, wkt, bkp, nullptr, nullptr, kb, Mtok, Dc, Dc);
  gemm_bf16_kernel<0><<<gDD, blk, 0, stream>>>(xb, wvt, bv,  nullptr, nullptr, vb, Mtok, Dc, Dc);

  const dim3 gAtt(Sc / 128, Bc * Hc);
  attn_kernel<<<gAtt, blk, 0, stream>>>(qb, kb, vb, rel, aob);

  gemm_bf16_kernel<1><<<gDD, blk, 0, stream>>>(aob, wot, bo, x, y1, nullptr, Mtok, Dc, Dc);
  rmsnorm_kernel<<<Mtok, blk, 0, stream>>>(y1, g1, y1n, y1nb, Dc);

  const dim3 gF1(DFFc / 256, Mtok / 128);
  gemm_bf16_kernel<2><<<gF1, blk, 0, stream>>>(y1nb, w1t, b1, nullptr, nullptr, hb, Mtok, DFFc, Dc);
  gemm_bf16_kernel<1><<<gDD, blk, 0, stream>>>(hb, w2t, b2, y1n, y1, nullptr, Mtok, Dc, DFFc);
  rmsnorm_kernel<<<Mtok, blk, 0, stream>>>(y1, g2, (float*)d_out, nullptr, Dc);
}
